// MultiHeadSelfAttention_12137577578825
// MI455X (gfx1250) — compile-verified, matrix operand order checked by reading, not run on hardware
//
#include <hip/hip_runtime.h>
#include <hip/hip_bf16.h>

#define L_SEQ   4096
#define DMODEL  1024
#define NHEADS  16
#define NKV     4
#define DH      64
#define DKV     256

typedef __attribute__((ext_vector_type(16))) __bf16 v16bf;
typedef __attribute__((ext_vector_type(8)))  __bf16 v8bf;
typedef __attribute__((ext_vector_type(8)))  float  v8f;

// ---------------------------------------------------------------------------
// CDNA5 async global->LDS path (ASYNCcnt), with safe fallback.
// Builtin signature (from hipcc diagnostic): pointers to 16B int vectors,
// global side in AS1, LDS side in AS3, then (offset, cpol).
// ---------------------------------------------------------------------------
#if defined(__AMDGCN__) && __has_builtin(__builtin_amdgcn_global_load_async_to_lds_b128)
#define HAVE_ASYNC_LDS 1
#else
#define HAVE_ASYNC_LDS 0
#endif

typedef int b128i __attribute__((vector_size(4 * sizeof(int))));

static __device__ __forceinline__ void async_cp16(const __bf16* g, __bf16* l) {
#if HAVE_ASYNC_LDS
  __builtin_amdgcn_global_load_async_to_lds_b128(
      (__attribute__((address_space(1))) b128i*)g,
      (__attribute__((address_space(3))) b128i*)l, 0, 0);
#else
  *reinterpret_cast<uint4*>(l) = *reinterpret_cast<const uint4*>(g);
#endif
}

#if HAVE_ASYNC_LDS
# if __has_builtin(__builtin_amdgcn_s_wait_asynccnt)
#  define ASYNC_WAIT0() __builtin_amdgcn_s_wait_asynccnt(0)
# else
#  define ASYNC_WAIT0() asm volatile("s_wait_asynccnt 0x0" ::: "memory")
# endif
#else
# define ASYNC_WAIT0() ((void)0)
#endif

// ---------------------------------------------------------------------------
// Fragment helpers. ISA 7.12.2 (wave32):
//   16-bit A (16x32 MxK): lane l holds row m=l&15; elements e=0..15 map to
//     k = (l>>4)*8 + (e&7) + (e>=8 ? 16 : 0)   -> two contiguous 16B chunks.
//   16-bit B (32x16 KxN): lane l holds col n=l&15 with identical k mapping.
//   f32 C/D (16x16): VGPR r holds m = r + 8*(l>>4), n = l&15.
// ---------------------------------------------------------------------------
static __device__ __forceinline__ v16bf load_frag(const __bf16* base, int ld) {
  const int lane = threadIdx.x & 31;
  const int r    = lane & 15;
  const int ko   = (lane >> 4) << 3;
  const __bf16* p = base + r * ld + ko;
  v8bf lo = *reinterpret_cast<const v8bf*>(p);
  v8bf hi = *reinterpret_cast<const v8bf*>(p + 16);
  v16bf out;
#pragma unroll
  for (int i = 0; i < 8; ++i) { out[i] = lo[i]; out[i + 8] = hi[i]; }
  return out;
}

static __device__ __forceinline__ v8f wmma_bf16(v16bf a, v16bf b, v8f c) {
  return __builtin_amdgcn_wmma_f32_16x16x32_bf16(false, a, false, b,
                                                 (short)0, c, false, false);
}

static __device__ __forceinline__ v8f v8f_zero() {
  v8f z = {0.f, 0.f, 0.f, 0.f, 0.f, 0.f, 0.f, 0.f};
  return z;
}

// ---------------------------------------------------------------------------
// fp32 -> bf16 elementwise convert
// ---------------------------------------------------------------------------
__global__ void cvt_f32_bf16(const float* __restrict__ src,
                             __bf16* __restrict__ dst, int n) {
  int i = blockIdx.x * blockDim.x + threadIdx.x;
  if (i < n) dst[i] = (__bf16)src[i];
}

// ---------------------------------------------------------------------------
// C[M,N] = A[M,K] * W[N,K]^T   (bf16 in, fp32 out)
// grid (N/128, M/128), 256 threads = 8 waves; wave (wm in 0..1, wn in 0..3)
// computes a 64x32 output patch as 4x2 WMMA accumulators.
// Double-buffered LDS; next K-chunk streamed via async global->LDS while
// the current chunk feeds the WMMAs.
// ---------------------------------------------------------------------------
__global__ __launch_bounds__(256)
void gemm_nt(const __bf16* __restrict__ A, const __bf16* __restrict__ W,
             float* __restrict__ C, int M, int N, int K) {
  __shared__ __attribute__((aligned(16))) __bf16 As[2][128 * 32];
  __shared__ __attribute__((aligned(16))) __bf16 Ws[2][128 * 32];
  const int tid  = threadIdx.x;
  const int wave = tid >> 5;
  const int lane = tid & 31;
  const int wm   = wave >> 2;
  const int wn   = wave & 3;
  const size_t m0 = (size_t)blockIdx.y * 128;
  const size_t n0 = (size_t)blockIdx.x * 128;

  // per-thread 16B-chunk coordinates (512 chunks per tile pair)
  const int c0 = tid,        r0 = c0 >> 2, col0 = (c0 & 3) << 3;
  const int c1 = tid + 256,  r1 = c1 >> 2, col1 = (c1 & 3) << 3;

  auto issue = [&](int kc, int buf) {
    async_cp16(&A[(m0 + r0) * K + kc + col0], &As[buf][c0 << 3]);
    async_cp16(&W[(n0 + r0) * K + kc + col0], &Ws[buf][c0 << 3]);
    async_cp16(&A[(m0 + r1) * K + kc + col1], &As[buf][c1 << 3]);
    async_cp16(&W[(n0 + r1) * K + kc + col1], &Ws[buf][c1 << 3]);
  };

  v8f acc[4][2];
#pragma unroll
  for (int i = 0; i < 4; ++i)
#pragma unroll
    for (int j = 0; j < 2; ++j) acc[i][j] = v8f_zero();

  issue(0, 0);
  int cur = 0;
  for (int kc = 0; kc < K; kc += 32) {
    ASYNC_WAIT0();                       // own wave's async copies landed
    __syncthreads();                     // all waves' copies visible
    if (kc + 32 < K) issue(kc + 32, cur ^ 1);

    v16bf b0 = load_frag(&Ws[cur][(wn * 32 + 0) * 32], 32);
    v16bf b1 = load_frag(&Ws[cur][(wn * 32 + 16) * 32], 32);
#pragma unroll
    for (int i = 0; i < 4; ++i) {
      v16bf a = load_frag(&As[cur][(wm * 64 + i * 16) * 32], 32);
      acc[i][0] = wmma_bf16(a, b0, acc[i][0]);
      acc[i][1] = wmma_bf16(a, b1, acc[i][1]);
    }
    cur ^= 1;
  }

  const int nloc = lane & 15;
  const int mhi  = (lane >> 4) << 3;
#pragma unroll
  for (int i = 0; i < 4; ++i)
#pragma unroll
    for (int j = 0; j < 2; ++j)
#pragma unroll
      for (int r = 0; r < 8; ++r)
        C[(m0 + wm * 64 + i * 16 + mhi + r) * N +
          n0 + wn * 32 + j * 16 + nloc] = acc[i][j][r];
}

// ---------------------------------------------------------------------------
// RoPE + repack kernels
// ---------------------------------------------------------------------------
__global__ void rope_pack_q(const float* __restrict__ Qf,
                            const int* __restrict__ pos,
                            __bf16* __restrict__ Qb) {
  int idx = blockIdx.x * blockDim.x + threadIdx.x;   // L * H * (DH/2)
  if (idx >= L_SEQ * NHEADS * (DH / 2)) return;
  int i = idx & 31;
  int h = (idx >> 5) & (NHEADS - 1);
  int l = idx >> 9;
  float p  = (float)pos[l];
  float fr = __powf(10000.0f, -(2.0f * i) / (float)DH);
  float s, c;
  __sincosf(p * fr, &s, &c);
  float ev = Qf[(size_t)l * DMODEL + h * DH + 2 * i];
  float od = Qf[(size_t)l * DMODEL + h * DH + 2 * i + 1];
  __bf16* dst = Qb + ((size_t)h * L_SEQ + l) * DH + 2 * i;
  dst[0] = (__bf16)(ev * c - od * s);
  dst[1] = (__bf16)(ev * s + od * c);
}

__global__ void rope_pack_k(const float* __restrict__ Kf,
                            const int* __restrict__ pos,
                            __bf16* __restrict__ Kb) {
  int idx = blockIdx.x * blockDim.x + threadIdx.x;   // L * NKV * (DH/2)
  if (idx >= L_SEQ * NKV * (DH / 2)) return;
  int i = idx & 31;
  int g = (idx >> 5) & (NKV - 1);
  int l = idx >> 7;
  float p  = (float)pos[l];
  float fr = __powf(10000.0f, -(2.0f * i) / (float)DH);
  float s, c;
  __sincosf(p * fr, &s, &c);
  float ev = Kf[(size_t)l * DKV + g * DH + 2 * i];
  float od = Kf[(size_t)l * DKV + g * DH + 2 * i + 1];
  __bf16* dst = Kb + ((size_t)g * L_SEQ + l) * DH + 2 * i;
  dst[0] = (__bf16)(ev * c - od * s);
  dst[1] = (__bf16)(ev * s + od * c);
}

// V repacked dim-major: Vb[g][d][l]  (so P*V B-fragments are contiguous)
__global__ void pack_v(const float* __restrict__ Vf, __bf16* __restrict__ Vb) {
  int idx = blockIdx.x * blockDim.x + threadIdx.x;   // L * DKV
  if (idx >= L_SEQ * DKV) return;
  int col = idx & (DKV - 1);
  int l   = idx >> 8;
  int g   = col >> 6;
  int dd  = col & 63;
  Vb[((size_t)g * DH + dd) * L_SEQ + l] = (__bf16)Vf[idx];
}

// ---------------------------------------------------------------------------
// Flash attention (causal, GQA g = h>>2).
// grid (L/128, H); 8 waves, each wave owns 16 query rows; 32-key chunks.
// K/V chunks double-buffered through LDS via async global->LDS.
// ---------------------------------------------------------------------------
__global__ __launch_bounds__(256)
void flash_attn(const __bf16* __restrict__ Qb, const __bf16* __restrict__ Kb,
                const __bf16* __restrict__ Vb, __bf16* __restrict__ Ob) {
  __shared__ __attribute__((aligned(16))) __bf16 Ks[2][32 * 64];  // key-major
  __shared__ __attribute__((aligned(16))) __bf16 Vs[2][64 * 32];  // dim-major
  __shared__ __attribute__((aligned(16))) __bf16 Ps[8][16 * 32];
  const int tid  = threadIdx.x;
  const int wave = tid >> 5;
  const int lane = tid & 31;
  const int h    = blockIdx.y;
  const int g    = h >> 2;
  const int q0   = blockIdx.x * 128;
  const int qw   = q0 + wave * 16;
  const int nloc = lane & 15;
  const int mhi  = (lane >> 4) << 3;
  const float scale = 0.125f;            // 1/sqrt(64)

  const int rk = tid >> 3, ck = (tid & 7) << 3;   // K chunk: 32x64
  const int rv = tid >> 2, cv = (tid & 3) << 3;   // V chunk: 64x32 (dim-major)
  auto issueKV = [&](int kb, int buf) {
    async_cp16(&Kb[((size_t)g * L_SEQ + kb + rk) * DH + ck], &Ks[buf][tid << 3]);
    async_cp16(&Vb[((size_t)g * DH + rv) * L_SEQ + kb + cv], &Vs[buf][tid << 3]);
  };

  const __bf16* Qh = Qb + ((size_t)h * L_SEQ + qw) * DH;
  v16bf aq0 = load_frag(Qh, DH);         // d = 0..31
  v16bf aq1 = load_frag(Qh + 32, DH);    // d = 32..63

  v8f o[4];
#pragma unroll
  for (int t = 0; t < 4; ++t) o[t] = v8f_zero();
  float mrow[8], lrow[8];
#pragma unroll
  for (int r = 0; r < 8; ++r) { mrow[r] = -3.0e38f; lrow[r] = 0.0f; }

  const int nk = q0 + 128;               // causal upper bound for this block
  issueKV(0, 0);
  int cur = 0;
  for (int kb = 0; kb < nk; kb += 32) {
    ASYNC_WAIT0();
    __syncthreads();
    if (kb + 32 < nk) issueKV(kb + 32, cur ^ 1);

    // S(16x32) = Q(16x64) * K^T : 4 WMMAs (2 key sub-tiles x 2 d-chunks)
    v8f s0 = v8f_zero(), s1 = v8f_zero();
    s0 = wmma_bf16(aq0, load_frag(&Ks[cur][0], 64), s0);
    s0 = wmma_bf16(aq1, load_frag(&Ks[cur][32], 64), s0);
    s1 = wmma_bf16(aq0, load_frag(&Ks[cur][16 * 64], 64), s1);
    s1 = wmma_bf16(aq1, load_frag(&Ks[cur][16 * 64 + 32], 64), s1);

    // online softmax; rows live in 16-lane halves -> shfl_xor 8..1
#pragma unroll
    for (int r = 0; r < 8; ++r) {
      int qrow = qw + mhi + r;
      float a = s0[r] * scale; if (kb + nloc > qrow)      a = -3.0e38f;
      float b = s1[r] * scale; if (kb + 16 + nloc > qrow) b = -3.0e38f;
      float mx = fmaxf(a, b);
#pragma unroll
      for (int off = 8; off > 0; off >>= 1)
        mx = fmaxf(mx, __shfl_xor(mx, off, 32));
      float nm   = fmaxf(mrow[r], mx);
      float pa   = __expf(a - nm);
      float pb   = __expf(b - nm);
      float corr = __expf(mrow[r] - nm);
      float rs   = pa + pb;
#pragma unroll
      for (int off = 8; off > 0; off >>= 1)
        rs += __shfl_xor(rs, off, 32);
      lrow[r] = lrow[r] * corr + rs;
      mrow[r] = nm;
#pragma unroll
      for (int t = 0; t < 4; ++t) o[t][r] *= corr;
      // restage P (C-layout) into row-major LDS for A-fragment reload
      Ps[wave][(mhi + r) * 32 + nloc]      = (__bf16)pa;
      Ps[wave][(mhi + r) * 32 + 16 + nloc] = (__bf16)pb;
    }
    // O(16x64) += P(16x32) * V(32x64) : 4 WMMAs (per-wave LDS, DS in-order)
    v16bf pf = load_frag(&Ps[wave][0], 32);
#pragma unroll
    for (int t = 0; t < 4; ++t)
      o[t] = wmma_bf16(pf, load_frag(&Vs[cur][t * 16 * 32], 32), o[t]);
    cur ^= 1;
  }

#pragma unroll
  for (int t = 0; t < 4; ++t)
#pragma unroll
    for (int r = 0; r < 8; ++r)
      Ob[(size_t)(qw + mhi + r) * DMODEL + h * DH + t * 16 + nloc] =
          (__bf16)(o[t][r] / lrow[r]);
}

// ---------------------------------------------------------------------------
// Host-side orchestration
// ---------------------------------------------------------------------------
extern "C" void kernel_launch(void* const* d_in, const int* in_sizes, int n_in,
                              void* d_out, int out_size, void* d_ws,
                              size_t ws_size, hipStream_t stream) {
  (void)in_sizes; (void)n_in; (void)out_size; (void)ws_size;
  const float* x  = (const float*)d_in[0];
  const float* Wq = (const float*)d_in[1];
  const float* Wk = (const float*)d_in[2];
  const float* Wv = (const float*)d_in[3];
  const float* Wo = (const float*)d_in[4];
  const int*   tp = (const int*)d_in[5];

  char* ws = (char*)d_ws;
  __bf16* xb  = (__bf16*)(ws + 0);          //  8 MiB  x bf16 [L][D]
  __bf16* Wqb = (__bf16*)(ws + 8388608);    //  2 MiB
  __bf16* Wkb = (__bf16*)(ws + 10485760);   //  512 KiB
  __bf16* Wvb = (__bf16*)(ws + 11010048);   //  512 KiB
  __bf16* Wob = (__bf16*)(ws + 11534336);   //  2 MiB
  float*  Qf  = (float*)(ws + 13631488);    // 16 MiB  fp32 [L][1024]
  float*  Kf  = (float*)(ws + 30408704);    //  4 MiB  fp32 [L][256]
  float*  Vf  = (float*)(ws + 34603008);    //  4 MiB
  __bf16* Qb  = (__bf16*)(ws + 38797312);   //  8 MiB  [H][L][64]
  __bf16* Kb  = (__bf16*)(ws + 47185920);   //  2 MiB  [Hkv][L][64]
  __bf16* Vb  = (__bf16*)(ws + 49283072);   //  2 MiB  [Hkv][64][L]
  __bf16* Ob  = (__bf16*)(ws + 51380224);   //  8 MiB  [L][1024]

  cvt_f32_bf16<<<(L_SEQ * DMODEL) / 256, 256, 0, stream>>>(x, xb, L_SEQ * DMODEL);
  cvt_f32_bf16<<<(DMODEL * DMODEL) / 256, 256, 0, stream>>>(Wq, Wqb, DMODEL * DMODEL);
  cvt_f32_bf16<<<(DKV * DMODEL) / 256, 256, 0, stream>>>(Wk, Wkb, DKV * DMODEL);
  cvt_f32_bf16<<<(DKV * DMODEL) / 256, 256, 0, stream>>>(Wv, Wvb, DKV * DMODEL);
  cvt_f32_bf16<<<(DMODEL * DMODEL) / 256, 256, 0, stream>>>(Wo, Wob, DMODEL * DMODEL);

  gemm_nt<<<dim3(8, 32), 256, 0, stream>>>(xb, Wqb, Qf, L_SEQ, DMODEL, DMODEL);
  gemm_nt<<<dim3(2, 32), 256, 0, stream>>>(xb, Wkb, Kf, L_SEQ, DKV, DMODEL);
  gemm_nt<<<dim3(2, 32), 256, 0, stream>>>(xb, Wvb, Vf, L_SEQ, DKV, DMODEL);

  rope_pack_q<<<(L_SEQ * NHEADS * 32) / 256, 256, 0, stream>>>(Qf, tp, Qb);
  rope_pack_k<<<(L_SEQ * NKV * 32) / 256, 256, 0, stream>>>(Kf, tp, Kb);
  pack_v<<<(L_SEQ * DKV) / 256, 256, 0, stream>>>(Vf, Vb);

  flash_attn<<<dim3(32, 16), 256, 0, stream>>>(Qb, Kb, Vb, Ob);

  gemm_nt<<<dim3(8, 32), 256, 0, stream>>>(Ob, Wob, (float*)d_out,
                                           L_SEQ, DMODEL, DMODEL);
}
